// BERT_20315195310332
// MI455X (gfx1250) — matrix operand order repaired and checked by reading, not compiled, not run on hardware
//
#include <hip/hip_runtime.h>
#include <math.h>

// ---------------------------------------------------------------------------
// BERT-base (L=12, T=512, D=768, H=12, F=3072) + CRF Viterbi for gfx1250.
// GEMMs: v_wmma_f32_16x16x32_bf16, register-blocked 2x2 tiles (32x32/wave)
// so each A/B fragment is reused twice (32 FLOP/byte from cache vs 16).
// Weights are converted fp32 -> bf16 TRANSPOSED via an LDS-tiled transpose
// (coalesced on both sides) so every WMMA B-fragment load is a contiguous
// b128 read of W^T rows.
// ---------------------------------------------------------------------------

typedef __bf16 bf16;
typedef __attribute__((ext_vector_type(16))) __bf16        v16bf;
typedef __attribute__((ext_vector_type(8)))  float         v8f;
typedef __attribute__((ext_vector_type(4)))  unsigned int  v4u;

#define T_SEQ   512
#define DMODEL  768
#define NHEAD   12
#define DHEAD   64
#define FDIM    3072
#define NLAYER  12
#define NTAGS   10
#define START_TAG 8
#define STOP_TAG  9
#define NEG_TAG  -10000.0f

__device__ __forceinline__ bf16 f2bf(float f) {
  unsigned u = __builtin_bit_cast(unsigned, f);
  u += 0x7FFFu + ((u >> 16) & 1u);            // round-to-nearest-even
  unsigned short h = (unsigned short)(u >> 16);
  return __builtin_bit_cast(bf16, h);
}

union FragU { v16bf v; v4u u[2]; };

#define WMMA_BF16(A_, B_, C_) \
  __builtin_amdgcn_wmma_f32_16x16x32_bf16(false, (A_), false, (B_), (short)0, (C_), false, false)

// ---------------------------------------------------------------------------
// Batched WMMA GEMM, 2x2 register blocking (32x32 output per wave):
//   C[b] = act( A[b] @ Bt[b]^T * scale + bias ) + res
//   A  : row-major [M,K] bf16, row stride lda, batch stride strideA
//   Bt : row-major [N,K] bf16 (B pre-transposed), ldb/strideB
//   C  : optional f32 [M,N];  obf: optional bf16 mirror; obfT: bf16 transposed
// K multiple of 32; M,N multiples of 32. Whole-wave-uniform early exit keeps
// EXEC all-ones at every WMMA (ISA 7.12 requirement).
// ---------------------------------------------------------------------------
__global__ void gemm_bf16_kernel(
    const bf16* __restrict__ A,  int lda,  long strideA,
    const bf16* __restrict__ Bt, int ldb,  long strideB,
    float* __restrict__ C,       int ldc,  long strideC,
    const float* __restrict__ bias,
    const float* __restrict__ resid, int ldr,
    bf16* __restrict__ obf,      int ldo,  long strideO,
    bf16* __restrict__ obfT,     int ldot, long strideOT,
    int M, int N, int K, float scale, int gelu)
{
  const int wave = threadIdx.x >> 5;
  const int lane = threadIdx.x & 31;
  const int tilesX = N >> 5;                  // 32-wide macro tiles
  const int nTiles = tilesX * (M >> 5);
  const int tile = blockIdx.x * (blockDim.x >> 5) + wave;
  if (tile >= nTiles) return;                 // whole-wave uniform exit
  const long b  = blockIdx.y;
  const int tm = tile / tilesX;
  const int tn = tile - tm * tilesX;
  const int m0 = tm << 5, n0 = tn << 5;
  const int half = lane >> 4;                 // 0: lanes 0-15, 1: lanes 16-31
  const int r    = lane & 15;

  // A frag (16x32 bf16): lane<16 -> row r, K 0..15; lane>=16 -> K 16..31
  const bf16* a0 = A  + b * strideA + (long)(m0 + r)      * lda + half * 16;
  const bf16* a1 = A  + b * strideA + (long)(m0 + 16 + r) * lda + half * 16;
  // B frag from Bt rows (16 contiguous bf16/lane): lane<16 -> K 0..15, lane>=16 -> K 16..31
  const bf16* b0 = Bt + b * strideB + (long)(n0 + r)      * ldb + half * 16;
  const bf16* b1 = Bt + b * strideB + (long)(n0 + 16 + r) * ldb + half * 16;

  v8f acc00 = {}, acc01 = {}, acc10 = {}, acc11 = {};
  for (int k = 0; k < K; k += 32) {
    FragU fa0, fa1, fb0, fb1;
    fa0.u[0] = *(const v4u*)(a0 + k);  fa0.u[1] = *(const v4u*)(a0 + k + 8);
    fa1.u[0] = *(const v4u*)(a1 + k);  fa1.u[1] = *(const v4u*)(a1 + k + 8);
    fb0.u[0] = *(const v4u*)(b0 + k);  fb0.u[1] = *(const v4u*)(b0 + k + 8);
    fb1.u[0] = *(const v4u*)(b1 + k);  fb1.u[1] = *(const v4u*)(b1 + k + 8);
    acc00 = WMMA_BF16(fa0.v, fb0.v, acc00);
    acc01 = WMMA_BF16(fa0.v, fb1.v, acc01);
    acc10 = WMMA_BF16(fa1.v, fb0.v, acc10);
    acc11 = WMMA_BF16(fa1.v, fb1.v, acc11);
  }

  // C/D layout per 16x16 tile: VGPR i -> (M = 8*half + i, N = lane%16)
  v8f accs[2][2] = {{acc00, acc01}, {acc10, acc11}};
  #pragma unroll
  for (int jm = 0; jm < 2; ++jm) {
    const int row0 = m0 + jm * 16 + half * 8;
    #pragma unroll
    for (int jn = 0; jn < 2; ++jn) {
      const int col = n0 + jn * 16 + r;
      const float bia = bias ? bias[col] : 0.0f;
      #pragma unroll
      for (int i = 0; i < 8; ++i) {
        float v = accs[jm][jn][i] * scale + bia;
        if (gelu) v = 0.5f * v * (1.0f + erff(v * 0.70710678118f)); // exact gelu
        const long rr = row0 + i;
        if (resid) v += resid[rr * (long)ldr + col];
        if (C)    C   [b * strideC  + rr * (long)ldc  + col] = v;
        if (obf)  obf [b * strideO  + rr * (long)ldo  + col] = f2bf(v);
        if (obfT) obfT[b * strideOT + (long)col * ldot + rr] = f2bf(v);
      }
    }
  }
}

// fp32 [K,N] -> bf16 [N,K] via 32x32 LDS tiles: coalesced reads AND writes.
// K, N must be multiples of 32 (true for 768 and 3072). Block (32,8).
__global__ void cvt_transpose_kernel(const float* __restrict__ W,
                                     bf16* __restrict__ Wt, int K, int N) {
  __shared__ bf16 tile[32][33];               // +1 pad: no LDS bank conflicts
  const int nb = blockIdx.x << 5;             // N-tile origin
  const int kb = blockIdx.y << 5;             // K-tile origin
  const int tx = threadIdx.x;                 // 0..31
  const int ty = threadIdx.y;                 // 0..7
  #pragma unroll
  for (int i = ty; i < 32; i += 8)            // read W[kb+i][nb+tx] (coalesced)
    tile[i][tx] = f2bf(W[(long)(kb + i) * N + (nb + tx)]);
  __syncthreads();
  #pragma unroll
  for (int i = ty; i < 32; i += 8)            // write Wt[nb+i][kb+tx] (coalesced)
    Wt[(long)(nb + i) * K + (kb + tx)] = tile[tx][i];
}

__global__ void embed_kernel(const int* __restrict__ sent, const float* __restrict__ we,
                             const float* __restrict__ pe, const float* __restrict__ te,
                             float* __restrict__ out) {
  const int t = blockIdx.x;
  const long w = sent[t];
  for (int d = threadIdx.x; d < DMODEL; d += blockDim.x)
    out[(long)t * DMODEL + d] = we[w * DMODEL + d] + pe[(long)t * DMODEL + d] + te[d];
}

// One wave per row of D=768; writes f32 + bf16 mirror.
__global__ void layernorm_kernel(const float* __restrict__ X, const float* __restrict__ g,
                                 const float* __restrict__ b, float* __restrict__ Y,
                                 bf16* __restrict__ Ybf) {
  const int row = blockIdx.x;
  const int lane = threadIdx.x;
  const float* x = X + (long)row * DMODEL;
  float vals[DMODEL / 32];
  float s = 0.f;
  #pragma unroll
  for (int i = 0; i < DMODEL / 32; ++i) { vals[i] = x[lane + i * 32]; s += vals[i]; }
  #pragma unroll
  for (int o = 16; o > 0; o >>= 1) s += __shfl_xor(s, o, 32);
  const float mean = s * (1.0f / DMODEL);
  float vv = 0.f;
  #pragma unroll
  for (int i = 0; i < DMODEL / 32; ++i) { float d = vals[i] - mean; vv += d * d; }
  #pragma unroll
  for (int o = 16; o > 0; o >>= 1) vv += __shfl_xor(vv, o, 32);
  const float rstd = rsqrtf(vv * (1.0f / DMODEL) + 1e-12f);
  #pragma unroll
  for (int i = 0; i < DMODEL / 32; ++i) {
    const int d = lane + i * 32;
    const float o = (vals[i] - mean) * rstd * g[d] + b[d];
    Y  [(long)row * DMODEL + d] = o;
    Ybf[(long)row * DMODEL + d] = f2bf(o);
  }
}

// One wave per score row (NHEAD*T rows, len T); emits bf16 probabilities.
__global__ void softmax_kernel(const float* __restrict__ S, bf16* __restrict__ P) {
  const int row = blockIdx.x;
  const int lane = threadIdx.x;
  const float* x = S + (long)row * T_SEQ;
  bf16* p = P + (long)row * T_SEQ;
  float vals[T_SEQ / 32];
  float mx = -INFINITY;
  #pragma unroll
  for (int i = 0; i < T_SEQ / 32; ++i) { vals[i] = x[lane + i * 32]; mx = fmaxf(mx, vals[i]); }
  #pragma unroll
  for (int o = 16; o > 0; o >>= 1) mx = fmaxf(mx, __shfl_xor(mx, o, 32));
  float s = 0.f;
  #pragma unroll
  for (int i = 0; i < T_SEQ / 32; ++i) { vals[i] = __expf(vals[i] - mx); s += vals[i]; }
  #pragma unroll
  for (int o = 16; o > 0; o >>= 1) s += __shfl_xor(s, o, 32);
  const float inv = 1.0f / s;
  #pragma unroll
  for (int i = 0; i < T_SEQ / 32; ++i) p[lane + i * 32] = f2bf(vals[i] * inv);
}

// feats = h @ fc_w + fc_b   (N=10 < 16, so plain f32 dot products)
__global__ void fc_kernel(const float* __restrict__ h, const float* __restrict__ w,
                          const float* __restrict__ bias, float* __restrict__ feats) {
  const int t = blockIdx.x;
  const int tag = threadIdx.x;
  if (tag >= NTAGS) return;
  float s = bias[tag];
  for (int d = 0; d < DMODEL; ++d) s += h[(long)t * DMODEL + d] * w[d * NTAGS + tag];
  feats[t * NTAGS + tag] = s;
}

// Sequential Viterbi (10 tags): one lane; backpointers in LDS.
// out[0] = path score; out[1..512] = tag path (as floats).
__global__ void viterbi_kernel(const float* __restrict__ feats,
                               const float* __restrict__ trans,
                               float* __restrict__ out) {
  __shared__ int bp[T_SEQ * NTAGS];
  if (threadIdx.x != 0) return;
  float tr[NTAGS * NTAGS];
  for (int i = 0; i < NTAGS * NTAGS; ++i) tr[i] = trans[i];
  float fv[NTAGS];
  for (int i = 0; i < NTAGS; ++i) fv[i] = (i == START_TAG) ? 0.0f : NEG_TAG;
  for (int t = 0; t < T_SEQ; ++t) {
    float nf[NTAGS];
    for (int n = 0; n < NTAGS; ++n) {
      float best = -INFINITY; int bi = 0;
      for (int p = 0; p < NTAGS; ++p) {
        float sc = fv[p] + tr[n * NTAGS + p];     // scores[next, prev]
        if (sc > best) { best = sc; bi = p; }
      }
      nf[n] = best + feats[t * NTAGS + n];
      bp[t * NTAGS + n] = bi;
    }
    for (int n = 0; n < NTAGS; ++n) fv[n] = nf[n];
  }
  float best = -INFINITY; int last = 0;
  for (int n = 0; n < NTAGS; ++n) {
    float sc = fv[n] + tr[STOP_TAG * NTAGS + n];
    if (sc > best) { best = sc; last = n; }
  }
  out[0] = best;
  int c = last;
  for (int t = T_SEQ - 1; t >= 0; --t) { out[1 + t] = (float)c; c = bp[t * NTAGS + c]; }
}

// ---------------------------------------------------------------------------
extern "C" void kernel_launch(void* const* d_in, const int* in_sizes, int n_in,
                              void* d_out, int out_size, void* d_ws, size_t ws_size,
                              hipStream_t stream) {
  (void)in_sizes; (void)n_in; (void)out_size; (void)ws_size;
  const int*   sent = (const int*)  d_in[0];
  const float* wemb = (const float*)d_in[1];
  const float* pemb = (const float*)d_in[2];
  const float* temb = (const float*)d_in[3];
  const float* elng = (const float*)d_in[4];
  const float* elnb = (const float*)d_in[5];
  const float* Wq = (const float*)d_in[6];  const float* bq = (const float*)d_in[7];
  const float* Wk = (const float*)d_in[8];  const float* bk = (const float*)d_in[9];
  const float* Wv = (const float*)d_in[10]; const float* bv = (const float*)d_in[11];
  const float* Wo = (const float*)d_in[12]; const float* bo = (const float*)d_in[13];
  const float* ln1g = (const float*)d_in[14]; const float* ln1b = (const float*)d_in[15];
  const float* W1 = (const float*)d_in[16]; const float* b1 = (const float*)d_in[17];
  const float* W2 = (const float*)d_in[18]; const float* b2 = (const float*)d_in[19];
  const float* ln2g = (const float*)d_in[20]; const float* ln2b = (const float*)d_in[21];
  const float* fcw = (const float*)d_in[22]; const float* fcb = (const float*)d_in[23];
  const float* trans = (const float*)d_in[24];
  float* out = (float*)d_out;

  // ---- workspace carve-up (~34 MB total) ----
  char* ws = (char*)d_ws;
  size_t off = 0;
  auto take = [&](size_t bytes) -> char* {
    char* p = ws + off; off = (off + bytes + 255) & ~(size_t)255; return p;
  };
  float* hF    = (float*)take((size_t)T_SEQ * DMODEL * 4);          // hidden f32
  float* tmpF  = (float*)take((size_t)T_SEQ * DMODEL * 4);          // pre-LN scratch
  bf16*  hB    = (bf16*) take((size_t)T_SEQ * DMODEL * 2);          // hidden bf16
  bf16*  qB    = (bf16*) take((size_t)T_SEQ * DMODEL * 2);
  bf16*  kB    = (bf16*) take((size_t)T_SEQ * DMODEL * 2);
  bf16*  vT    = (bf16*) take((size_t)DMODEL * T_SEQ * 2);          // V^T [768,512]
  bf16*  ctxB  = (bf16*) take((size_t)T_SEQ * DMODEL * 2);
  bf16*  gB    = (bf16*) take((size_t)T_SEQ * FDIM * 2);            // gelu(FFN1)
  float* sc    = (float*)take((size_t)NHEAD * T_SEQ * T_SEQ * 4);   // scores
  bf16*  pr    = (bf16*) take((size_t)NHEAD * T_SEQ * T_SEQ * 2);   // softmax probs
  bf16*  wT    = (bf16*) take((size_t)FDIM * DMODEL * 2);           // W^T bf16 scratch
  float* feats = (float*)take((size_t)T_SEQ * NTAGS * 4);

  auto cvt = [&](const float* W, int K, int N) {
    dim3 grid((unsigned)(N / 32), (unsigned)(K / 32));
    cvt_transpose_kernel<<<grid, dim3(32, 8), 0, stream>>>(W, wT, K, N);
  };
  auto gemm = [&](const bf16* A, int lda, long sA, const bf16* B, int ldb, long sB,
                  float* C, int ldc, long sC, const float* bias,
                  const float* res, int ldr, bf16* o, int ldo, long sO,
                  bf16* oT, int ldot, long sOT, int M, int N, int K, int batch,
                  float scale, int gelu) {
    int tiles = (M >> 5) * (N >> 5);          // 32x32 macro tiles, 1 per wave
    dim3 grid((unsigned)((tiles + 3) / 4), (unsigned)batch);
    gemm_bf16_kernel<<<grid, 128, 0, stream>>>(A, lda, sA, B, ldb, sB, C, ldc, sC,
                                               bias, res, ldr, o, ldo, sO,
                                               oT, ldot, sOT, M, N, K, scale, gelu);
  };

  // ---- embeddings + LN ----
  embed_kernel<<<T_SEQ, 256, 0, stream>>>(sent, wemb, pemb, temb, tmpF);
  layernorm_kernel<<<T_SEQ, 32, 0, stream>>>(tmpF, elng, elnb, hF, hB);

  for (int l = 0; l < NLAYER; ++l) {
    const long wsz = (long)DMODEL * DMODEL;
    const float* Wq_l = Wq + (long)l * wsz; const float* bq_l = bq + (long)l * DMODEL;
    const float* Wk_l = Wk + (long)l * wsz; const float* bk_l = bk + (long)l * DMODEL;
    const float* Wv_l = Wv + (long)l * wsz; const float* bv_l = bv + (long)l * DMODEL;
    const float* Wo_l = Wo + (long)l * wsz; const float* bo_l = bo + (long)l * DMODEL;
    const float* W1_l = W1 + (long)l * DMODEL * FDIM; const float* b1_l = b1 + (long)l * FDIM;
    const float* W2_l = W2 + (long)l * FDIM * DMODEL; const float* b2_l = b2 + (long)l * DMODEL;

    // Q = h@Wq + bq  (bf16 mirror only)
    cvt(Wq_l, DMODEL, DMODEL);
    gemm(hB, DMODEL, 0, wT, DMODEL, 0, nullptr, 0, 0, bq_l, nullptr, 0,
         qB, DMODEL, 0, nullptr, 0, 0, T_SEQ, DMODEL, DMODEL, 1, 1.0f, 0);
    // K
    cvt(Wk_l, DMODEL, DMODEL);
    gemm(hB, DMODEL, 0, wT, DMODEL, 0, nullptr, 0, 0, bk_l, nullptr, 0,
         kB, DMODEL, 0, nullptr, 0, 0, T_SEQ, DMODEL, DMODEL, 1, 1.0f, 0);
    // V (stored transposed [768,512] so ctx GEMM B-loads are contiguous)
    cvt(Wv_l, DMODEL, DMODEL);
    gemm(hB, DMODEL, 0, wT, DMODEL, 0, nullptr, 0, 0, bv_l, nullptr, 0,
         nullptr, 0, 0, vT, T_SEQ, 0, T_SEQ, DMODEL, DMODEL, 1, 1.0f, 0);

    // scores[h] = Q_h @ K_h^T / 8   (batched over 12 heads)
    gemm(qB, DMODEL, DHEAD, kB, DMODEL, DHEAD,
         sc, T_SEQ, (long)T_SEQ * T_SEQ, nullptr, nullptr, 0,
         nullptr, 0, 0, nullptr, 0, 0, T_SEQ, T_SEQ, DHEAD, NHEAD, 0.125f, 0);
    softmax_kernel<<<NHEAD * T_SEQ, 32, 0, stream>>>(sc, pr);

    // ctx[h] = P_h @ V_h   (batched; bf16 mirror into [T,768] at head column)
    gemm(pr, T_SEQ, (long)T_SEQ * T_SEQ, vT, T_SEQ, (long)DHEAD * T_SEQ,
         nullptr, 0, 0, nullptr, nullptr, 0,
         ctxB, DMODEL, DHEAD, nullptr, 0, 0, T_SEQ, DHEAD, T_SEQ, NHEAD, 1.0f, 0);

    // attn_out = h + ctx@Wo + bo ;  LN1
    cvt(Wo_l, DMODEL, DMODEL);
    gemm(ctxB, DMODEL, 0, wT, DMODEL, 0, tmpF, DMODEL, 0, bo_l, hF, DMODEL,
         nullptr, 0, 0, nullptr, 0, 0, T_SEQ, DMODEL, DMODEL, 1, 1.0f, 0);
    layernorm_kernel<<<T_SEQ, 32, 0, stream>>>(tmpF, ln1g + (long)l * DMODEL,
                                               ln1b + (long)l * DMODEL, hF, hB);

    // g = gelu(h@W1 + b1)  (bf16 mirror only)
    cvt(W1_l, DMODEL, FDIM);
    gemm(hB, DMODEL, 0, wT, DMODEL, 0, nullptr, 0, 0, b1_l, nullptr, 0,
         gB, FDIM, 0, nullptr, 0, 0, T_SEQ, FDIM, DMODEL, 1, 1.0f, 1);
    // ffn_out = h + g@W2 + b2 ;  LN2
    cvt(W2_l, FDIM, DMODEL);
    gemm(gB, FDIM, 0, wT, FDIM, 0, tmpF, DMODEL, 0, b2_l, hF, DMODEL,
         nullptr, 0, 0, nullptr, 0, 0, T_SEQ, DMODEL, FDIM, 1, 1.0f, 0);
    layernorm_kernel<<<T_SEQ, 32, 0, stream>>>(tmpF, ln2g + (long)l * DMODEL,
                                               ln2b + (long)l * DMODEL, hF, hB);
  }

  fc_kernel<<<T_SEQ, 16, 0, stream>>>(hF, fcw, fcb, feats);
  viterbi_kernel<<<1, 32, 0, stream>>>(feats, trans, out);
}
